// GCNN_2138893713526
// MI455X (gfx1250) — matrix operand order repaired and checked by reading, not compiled, not run on hardware
//
#include <hip/hip_runtime.h>
#include <hip/hip_bf16.h>

typedef __attribute__((ext_vector_type(16))) __bf16          v16bf;
typedef __attribute__((ext_vector_type(16))) unsigned short  v16us;
typedef __attribute__((ext_vector_type(8)))  float           v8f;
typedef __attribute__((ext_vector_type(8)))  unsigned short  us8;
typedef __attribute__((ext_vector_type(4)))  unsigned int    v4u;
typedef __attribute__((ext_vector_type(8)))  int             v8i;
typedef __attribute__((ext_vector_type(4)))  int             v4i;

#ifndef __has_builtin
#define __has_builtin(x) 0
#endif
#if __has_builtin(__builtin_amdgcn_tensor_load_to_lds) && \
    __has_builtin(__builtin_amdgcn_s_wait_tensorcnt)
#define USE_TDM 1
#else
#define USE_TDM 0
#endif

#define DD     32
#define NTOK   2000
#define NLEFT  500
#define NSTATE 1500
#define SLOPE  0.2f
#define WPR    63            // bit-mask words per row: ceil(2000/32)

// ---- workspace layout (float words) ----
#define OFF_G    0          // 1,024,000
#define OFF_SL   1024000    //    32,000
#define OFF_SR   1056000    //    32,000
#define OFF_H    1088000    // 1,024,000
#define OFF_ST   2112000    // 1,024,000
#define OFF_AGG  3136000    // 1,024,000
#define OFF_MV   4160000    //       512
#define OFF_SV   4160512    //       512
#define F32WORDS 4161024
// ---- bf16 region (ushort offsets from bf base) ----
#define OFF_EBF   0         //   512,000  (E[16][16][2000])
#define OFF_W4BF  512000    //    64,000  (w4[32][2000])
#define OFF_STBF  576000    // 1,024,000  (st bf16)
#define US_WORDS  1600000
// ---- bit-mask region: 32000 rows * 63 u32 = 2,016,000 u32 (8.06 MB) ----

__device__ __forceinline__ unsigned short f2bf(float f) {
  unsigned int u = __builtin_bit_cast(unsigned int, f);
  u += 0x7FFFu + ((u >> 16) & 1u);      // round-to-nearest-even
  return (unsigned short)(u >> 16);
}
__device__ __forceinline__ float wave_sum(float v) {
  #pragma unroll
  for (int m = 16; m > 0; m >>= 1) v += __shfl_xor(v, m, 32);
  return v;
}
__device__ __forceinline__ float wave_max(float v) {
  #pragma unroll
  for (int m = 16; m > 0; m >>= 1) v = fmaxf(v, __shfl_xor(v, m, 32));
  return v;
}

// A-tile (GEMM1): av[e0+q] = bit(kb+q) ? E[j0+kb+q] : 0   (bf16)
__device__ __forceinline__ void sel_bitsE(unsigned word, const unsigned short* erow,
                                          int j0, int kb, bool in, v16us& av, int e0) {
  if (in) {
    us8 e = *(const us8*)(erow + j0 + kb);
    unsigned wb = word >> kb;
    #pragma unroll
    for (int q = 0; q < 8; ++q)
      av[e0+q] = ((wb >> q) & 1u) ? e[q] : (unsigned short)0;
  } else {
    #pragma unroll
    for (int q = 0; q < 8; ++q) av[e0+q] = 0;
  }
}
// A-tile (GEMM2): av[e0+q] = bit ? bf16(1.0) : 0  (tail bits already zero)
__device__ __forceinline__ void sel_bits1(unsigned word, int kb, v16us& av, int e0) {
  unsigned wb = word >> kb;
  #pragma unroll
  for (int q = 0; q < 8; ++q)
    av[e0+q] = ((wb >> q) & 1u) ? (unsigned short)0x3F80 : (unsigned short)0;
}
// B-tile loader: 16 contiguous bf16 K-values (K = kb..kb+15 for this lane half)
__device__ __forceinline__ void load_brow(const unsigned short* brow, int kb,
                                          bool in, v16us& bv) {
  if (in) {
    us8 b0 = *(const us8*)(brow + kb);
    us8 b1 = *(const us8*)(brow + kb + 8);
    #pragma unroll
    for (int q = 0; q < 8; ++q) { bv[q] = b0[q]; bv[8+q] = b1[q]; }
  } else {
    #pragma unroll
    for (int q = 0; q < 16; ++q) bv[q] = 0;
  }
}

// K0: pack f32 mask (256 MB, streamed once, non-temporal) -> 1 bit/elem (8 MB)
__global__ __launch_bounds__(256) void k0_pack(
    const float* __restrict__ mask, unsigned int* __restrict__ bits) {
  int wave = threadIdx.x >> 5, lane = threadIdx.x & 31;
  long long W = (long long)blockIdx.x * 8 + wave;   // 0 .. 2,015,999
  int row = (int)(W / WPR);                         // 0 .. 31999 (b*2000+i)
  int w   = (int)(W % WPR);
  int j = w * 32 + lane;
  float mval = 0.f;
  if (j < NTOK) mval = __builtin_nontemporal_load(mask + (size_t)row * NTOK + j);
  unsigned int word = __builtin_amdgcn_ballot_w32(mval != 0.f);
  if (lane == 0) bits[(size_t)row * WPR + w] = word;
}

// K1: g = x@w3^T + b3 ; s_l = g.a_l ; s_r = g.a_r     (one wave per row)
__global__ __launch_bounds__(256) void k1_embed(
    const float* __restrict__ state, const float* __restrict__ left,
    const float* __restrict__ w3, const float* __restrict__ b3,
    const float* __restrict__ attn_w,
    float* __restrict__ g, float* __restrict__ sl, float* __restrict__ sr) {
  __shared__ float xbuf[256];
  int wave = threadIdx.x >> 5, lane = threadIdx.x & 31;
  int row = blockIdx.x * 8 + wave;               // 0..31999
  int b = row / NTOK, i = row % NTOK;
  float xv = (i < NLEFT) ? left[((size_t)b * NLEFT + i) * DD + lane]
                         : state[((size_t)b * NSTATE + (i - NLEFT)) * DD + lane];
  xbuf[wave * 32 + lane] = xv;
  __syncthreads();
  float acc = b3[lane];
  #pragma unroll
  for (int j = 0; j < 32; ++j) acc += xbuf[wave * 32 + j] * w3[lane * 32 + j];
  g[(size_t)row * 32 + lane] = acc;
  float psl = wave_sum(acc * attn_w[lane]);
  float psr = wave_sum(acc * attn_w[32 + lane]);
  if (lane == 0) { sl[row] = psl; sr[row] = psr; }
}

// K2: E[b][k][j] = leakyrelu(s_l[b,j] + s_r[k,j]) in bf16 ; w4 -> bf16
__global__ __launch_bounds__(256) void k2_prep(
    const float* __restrict__ sl, const float* __restrict__ sr,
    const float* __restrict__ w4,
    unsigned short* __restrict__ ebf, unsigned short* __restrict__ w4bf) {
  int tid = blockIdx.x * 256 + threadIdx.x;      // 0..575999
  if (tid < 512000) {
    int b = tid / 32000, rem = tid % 32000;
    int k = rem / NTOK, j = rem % NTOK;
    float v = sl[b * NTOK + j] + sr[k * NTOK + j];
    v = v > 0.f ? v : SLOPE * v;
    ebf[tid] = f2bf(v);
  } else {
    int t = tid - 512000;                        // < 64000
    w4bf[t] = f2bf(w4[t]);
  }
}

// K3: h = leakyrelu( (mask .* E) @ w4^T + b4 )   via v_wmma_f32_16x16x32_bf16
__global__ __launch_bounds__(256) void k3_gemm1(
    const unsigned int* __restrict__ bits, const unsigned short* __restrict__ ebf,
    const unsigned short* __restrict__ w4bf, const float* __restrict__ b4,
    float* __restrict__ h) {
  __shared__ unsigned short Elds[16 * NTOK];     // 64000 B
  int b = blockIdx.y;
#if USE_TDM
  {
    if ((threadIdx.x >> 5) == 0) {   // wave 0 issues one TDM DMA: 64000 B -> LDS
      unsigned long long ga = (unsigned long long)(const void*)(ebf + (size_t)b * 16 * NTOK);
      unsigned ldsoff = (unsigned)(unsigned long long)(void*)&Elds[0];
      v4u g0; v8i g1; v4i g2 = {0, 0, 0, 0}; v4i g3 = {0, 0, 0, 0};
      // D# group0: count=1 | lds_addr | 57b global_addr | type=2 ("image")
      g0[0] = 1u;
      g0[1] = ldsoff;
      g0[2] = (unsigned)(ga & 0xFFFFFFFFu);
      g0[3] = (unsigned)((ga >> 32) & 0x01FFFFFFu) | 0x80000000u;
      // D# group1: data_size=1 (2B); 1-D: tensor_dim0 = tile_dim0 = 32000
      g1[0] = 0x00010000;                  // [17:16] data_size = 1
      g1[1] = (int)(32000u << 16);         // tensor_dim0[15:0]  @ bits 63:48
      g1[2] = 0;                           // tensor_dim0 hi / tensor_dim1 lo
      g1[3] = (int)(32000u << 16);         // tile_dim0          @ bits 127:112
      g1[4] = 0;                           // tile_dim1/2 = 0 (unused)
      g1[5] = 32000;                       // tensor_dim0_stride low 32
      g1[6] = 0;
      g1[7] = 0;
#if __clang_major__ >= 23
      v8i gx = {0, 0, 0, 0, 0, 0, 0, 0};   // clang-23 toolchain: 6-arg form
      __builtin_amdgcn_tensor_load_to_lds(g0, g1, g2, g3, gx, 0);
#else
      __builtin_amdgcn_tensor_load_to_lds(g0, g1, g2, g3, 0);
#endif
      __builtin_amdgcn_s_wait_tensorcnt(0);
    }
    __syncthreads();
  }
#else
  {
    const unsigned int* src = (const unsigned int*)(ebf + (size_t)b * 16 * NTOK);
    unsigned int* dst = (unsigned int*)Elds;
    for (int w = threadIdx.x; w < 16 * NTOK / 2; w += 256) dst[w] = src[w];
  }
  __syncthreads();
#endif
  int wave = threadIdx.x >> 5, lane = threadIdx.x & 31;
  int tt = blockIdx.x * 8 + wave;
  if (tt >= 125) return;
  int i0 = tt * 16;
  int r = lane & 15, hi = lane >> 4;
  int kbA0 = hi * 16, kbA1 = kbA0 + 8;           // A: K {0-15} / {16-31}
  int kbB  = hi * 16;                            // B: K {0-15} / {16-31}
  const unsigned int* browbits = bits + ((size_t)b * NTOK + (i0 + r)) * WPR;
  const unsigned short* erow = Elds + r * NTOK;
  const unsigned short* b0row = w4bf + (size_t)r * NTOK;
  const unsigned short* b1row = w4bf + (size_t)(16 + r) * NTOK;
  v8f acc0 = {}, acc1 = {};
  for (int jc = 0; jc < WPR; ++jc) {
    int j0 = jc * 32;
    unsigned int word = browbits[jc];
    v16us av, bv0, bv1;
    sel_bitsE(word, erow, j0, kbA0, (j0 + kbA0) < NTOK, av, 0);
    sel_bitsE(word, erow, j0, kbA1, (j0 + kbA1) < NTOK, av, 8);
    load_brow(b0row, j0 + kbB, (j0 + kbB) < NTOK, bv0);
    load_brow(b1row, j0 + kbB, (j0 + kbB) < NTOK, bv1);
    v16bf a = __builtin_bit_cast(v16bf, av);
    acc0 = __builtin_amdgcn_wmma_f32_16x16x32_bf16(
        false, a, false, __builtin_bit_cast(v16bf, bv0), (short)0, acc0, false, false);
    acc1 = __builtin_amdgcn_wmma_f32_16x16x32_bf16(
        false, a, false, __builtin_bit_cast(v16bf, bv1), (short)0, acc1, false, false);
  }
  float bo0 = b4[r], bo1 = b4[16 + r];
  #pragma unroll
  for (int rr = 0; rr < 8; ++rr) {
    int row = i0 + rr + hi * 8;                  // D layout: M = rr + 8*hi, N = r
    float v0 = acc0[rr] + bo0; v0 = v0 > 0.f ? v0 : SLOPE * v0;
    float v1 = acc1[rr] + bo1; v1 = v1 > 0.f ? v1 : SLOPE * v1;
    h[((size_t)b * NTOK + row) * 32 + r] = v0;
    h[((size_t)b * NTOK + row) * 32 + 16 + r] = v1;
  }
}

// K4: softmax stats over axis i per (b,o): max and sum(exp)
__global__ __launch_bounds__(256) void k4_stats(
    const float* __restrict__ h, float* __restrict__ mv, float* __restrict__ sv) {
  int wave = threadIdx.x >> 5, lane = threadIdx.x & 31;
  int p = blockIdx.x * 8 + wave;                 // 0..511
  int b = p >> 5, o = p & 31;
  const float* hp = h + (size_t)b * NTOK * 32 + o;
  float m = -3.4e38f;
  for (int i = lane; i < NTOK; i += 32) m = fmaxf(m, hp[(size_t)i * 32]);
  m = wave_max(m);
  float s = 0.f;
  for (int i = lane; i < NTOK; i += 32) s += expf(hp[(size_t)i * 32] - m);
  s = wave_sum(s);
  if (lane == 0) { mv[p] = m; sv[p] = s; }
}

// K5: st = (softmax(h) .* g) @ w1^T + b1 ; also st in bf16
__global__ __launch_bounds__(256) void k5_st(
    const float* __restrict__ h, const float* __restrict__ g,
    const float* __restrict__ mv, const float* __restrict__ sv,
    const float* __restrict__ w1, const float* __restrict__ b1,
    float* __restrict__ st, unsigned short* __restrict__ stbf) {
  __shared__ float pbuf[256];
  int wave = threadIdx.x >> 5, lane = threadIdx.x & 31;
  int row = blockIdx.x * 8 + wave;
  int b = row / NTOK;
  float hv = h[(size_t)row * 32 + lane];
  float a = expf(hv - mv[b * 32 + lane]) / sv[b * 32 + lane];
  pbuf[wave * 32 + lane] = a * g[(size_t)row * 32 + lane];
  __syncthreads();
  float acc = b1[lane];
  #pragma unroll
  for (int d = 0; d < 32; ++d) acc += pbuf[wave * 32 + d] * w1[lane * 32 + d];
  st[(size_t)row * 32 + lane] = acc;
  stbf[(size_t)row * 32 + lane] = f2bf(acc);
}

// K6: agg = mask @ st   via WMMA; st tile transposed through LDS per K-chunk
__global__ __launch_bounds__(256) void k6_gemm2(
    const unsigned int* __restrict__ bits, const unsigned short* __restrict__ stbf,
    float* __restrict__ agg) {
  __shared__ unsigned short stT[32 * 32];        // B tile: [n][k]
  int b = blockIdx.y;
  int wave = threadIdx.x >> 5, lane = threadIdx.x & 31;
  int tt = blockIdx.x * 8 + wave;
  bool active = tt < 125;
  int i0 = (active ? tt : 124) * 16;
  int r = lane & 15, hi = lane >> 4;
  int kbA0 = hi * 16, kbA1 = kbA0 + 8, kbB = hi * 16;
  const unsigned int* browbits = bits + ((size_t)b * NTOK + (i0 + r)) * WPR;
  v8f acc0 = {}, acc1 = {};
  for (int jc = 0; jc < WPR; ++jc) {
    int j0 = jc * 32;
    {   // cooperative transpose-stage of st[b, j0..j0+31, 0..31]
      int l = threadIdx.x * 4;
      int k = l >> 5, n = l & 31;
      int jrow = j0 + k;
      unsigned short q0 = 0, q1 = 0, q2 = 0, q3 = 0;
      if (jrow < NTOK) {
        const unsigned short* sp = stbf + ((size_t)b * NTOK + jrow) * 32 + n;
        q0 = sp[0]; q1 = sp[1]; q2 = sp[2]; q3 = sp[3];
      }
      stT[(n + 0) * 32 + k] = q0;
      stT[(n + 1) * 32 + k] = q1;
      stT[(n + 2) * 32 + k] = q2;
      stT[(n + 3) * 32 + k] = q3;
    }
    __syncthreads();
    if (active) {
      unsigned int word = browbits[jc];
      v16us av, bv0, bv1;
      sel_bits1(word, kbA0, av, 0);
      sel_bits1(word, kbA1, av, 8);
      load_brow(stT + r * 32, kbB, true, bv0);         // zero-padded already
      load_brow(stT + (16 + r) * 32, kbB, true, bv1);
      v16bf a = __builtin_bit_cast(v16bf, av);
      acc0 = __builtin_amdgcn_wmma_f32_16x16x32_bf16(
          false, a, false, __builtin_bit_cast(v16bf, bv0), (short)0, acc0, false, false);
      acc1 = __builtin_amdgcn_wmma_f32_16x16x32_bf16(
          false, a, false, __builtin_bit_cast(v16bf, bv1), (short)0, acc1, false, false);
    }
    __syncthreads();
  }
  if (active) {
    #pragma unroll
    for (int rr = 0; rr < 8; ++rr) {
      int row = i0 + rr + hi * 8;
      agg[((size_t)b * NTOK + row) * 32 + r] = acc0[rr];
      agg[((size_t)b * NTOK + row) * 32 + 16 + r] = acc1[rr];
    }
  }
}

// K7: gates/LSTM cell, layernorm, residual, out = (...) @ w2^T + b2
__global__ __launch_bounds__(256) void k7_final(
    const float* __restrict__ st, const float* __restrict__ agg,
    const float* __restrict__ w_ih, const float* __restrict__ b_ih,
    const float* __restrict__ b_hh, const float* __restrict__ ln_g,
    const float* __restrict__ ln_b, const float* __restrict__ w2,
    const float* __restrict__ b2, float* __restrict__ out) {
  __shared__ float buf[256];
  int wave = threadIdx.x >> 5, lane = threadIdx.x & 31;
  int row = blockIdx.x * 8 + wave;
  float stv = st[(size_t)row * 32 + lane];
  buf[wave * 32 + lane] = agg[(size_t)row * 32 + lane];
  __syncthreads();
  float gi = b_ih[lane] + b_hh[lane];
  float gf = b_ih[32 + lane] + b_hh[32 + lane];
  float gg = b_ih[64 + lane] + b_hh[64 + lane];
  #pragma unroll
  for (int d = 0; d < 32; ++d) {
    float ag = buf[wave * 32 + d];
    gi += ag * w_ih[lane * 32 + d];
    gf += ag * w_ih[(32 + lane) * 32 + d];
    gg += ag * w_ih[(64 + lane) * 32 + d];
  }
  float ig  = 1.f / (1.f + expf(-gi));
  float fg  = 1.f / (1.f + expf(-gf));
  float g_g = tanhf(gg);
  float mean = wave_sum(stv) * (1.f / 32.f);
  float diff = stv - mean;
  float var  = wave_sum(diff * diff) * (1.f / 32.f);
  float nx = ln_g[lane] * diff * rsqrtf(var + 1e-6f) + ln_b[lane];
  float outv = stv + (fg * nx + ig * g_g);
  __syncthreads();
  buf[wave * 32 + lane] = outv;
  __syncthreads();
  float res = b2[lane];
  #pragma unroll
  for (int d = 0; d < 32; ++d) res += buf[wave * 32 + d] * w2[lane * 32 + d];
  out[(size_t)row * 32 + lane] = res;
}

extern "C" void kernel_launch(void* const* d_in, const int* in_sizes, int n_in,
                              void* d_out, int out_size, void* d_ws, size_t ws_size,
                              hipStream_t stream) {
  const float* state  = (const float*)d_in[0];
  const float* left   = (const float*)d_in[1];
  const float* mask   = (const float*)d_in[2];
  const float* w3     = (const float*)d_in[3];
  const float* b3     = (const float*)d_in[4];
  const float* attn_w = (const float*)d_in[5];
  const float* w4     = (const float*)d_in[6];
  const float* b4     = (const float*)d_in[7];
  const float* w1     = (const float*)d_in[8];
  const float* b1     = (const float*)d_in[9];
  const float* w_ih   = (const float*)d_in[10];
  const float* b_ih   = (const float*)d_in[12];
  const float* b_hh   = (const float*)d_in[13];
  const float* ln_g   = (const float*)d_in[14];
  const float* ln_b   = (const float*)d_in[15];
  const float* w2     = (const float*)d_in[16];
  const float* b2     = (const float*)d_in[17];

  float* f   = (float*)d_ws;
  float* g   = f + OFF_G;
  float* sl  = f + OFF_SL;
  float* sr  = f + OFF_SR;
  float* h   = f + OFF_H;
  float* st  = f + OFF_ST;
  float* agg = f + OFF_AGG;
  float* mv  = f + OFF_MV;
  float* sv  = f + OFF_SV;
  unsigned short* us   = (unsigned short*)(f + F32WORDS);
  unsigned short* ebf  = us + OFF_EBF;
  unsigned short* w4bf = us + OFF_W4BF;
  unsigned short* stbf = us + OFF_STBF;
  unsigned int*   bits = (unsigned int*)(us + US_WORDS);   // 2,016,000 u32
  float* out = (float*)d_out;

  k0_pack  <<<252000, 256, 0, stream>>>(mask, bits);
  k1_embed <<<4000, 256, 0, stream>>>(state, left, w3, b3, attn_w, g, sl, sr);
  k2_prep  <<<2250, 256, 0, stream>>>(sl, sr, w4, ebf, w4bf);
  k3_gemm1 <<<dim3(16, 16), 256, 0, stream>>>(bits, ebf, w4bf, b4, h);
  k4_stats <<<64, 256, 0, stream>>>(h, mv, sv);
  k5_st    <<<4000, 256, 0, stream>>>(h, g, mv, sv, w1, b1, st, stbf);
  k6_gemm2 <<<dim3(16, 16), 256, 0, stream>>>(bits, stbf, agg);
  k7_final <<<4000, 256, 0, stream>>>(st, agg, w_ih, b_ih, b_hh, ln_g, ln_b, w2, b2, out);
}
